// LinearAttention_23613730193858
// MI455X (gfx1250) — matrix operand order repaired and checked by reading, not compiled, not run on hardware
//
#include <hip/hip_runtime.h>

// ---------------------------------------------------------------------------
// Linear attention on gfx1250 via v_wmma_f32_16x16x32_bf16.
// All GEMMs are NT (both operands K-major contiguous) so WMMA fragments are
// plain contiguous 16B loads. wave32; block = 128 threads = 4 waves.
// k-loops are fully unrolled (compile-time trip counts) so fragments are pure
// SSA values: no loop-carried phis -> no accumulator rotation copies; the
// scheduler pipelines load batches across steps automatically.
// ---------------------------------------------------------------------------

typedef __bf16 bf16_t;
typedef __attribute__((ext_vector_type(16))) __bf16 v16bf;
typedef __attribute__((ext_vector_type(8)))  __bf16 v8bf;
typedef __attribute__((ext_vector_type(8)))  float  v8f;

#define B_  2
#define T_  4096
#define D_  1024
#define H_  16
#define QK_ 64
#define VD_ 64

// ---- WMMA fragment loaders (layouts per CDNA5 ISA 7.12.2) ------------------

// A-matrix 16x32 bf16, row-major source, tile origin at A, row stride lda.
// lane<16 : row=lane,    elems 0..7 = K 0..7,   elems 8..15 = K 16..23
// lane>=16: row=lane-16, elems 0..7 = K 8..15,  elems 8..15 = K 24..31
__device__ __forceinline__ v16bf load_a_frag(const bf16_t* __restrict__ A, int lda) {
  const int lane = threadIdx.x & 31;
  const int half = lane >> 4;
  const int r    = lane & 15;
  const bf16_t* base = A + (size_t)r * lda;
  v8bf lo = *(const v8bf*)(base + 8 * half);
  v8bf hi = *(const v8bf*)(base + 16 + 8 * half);
  v16bf out;
#pragma unroll
  for (int i = 0; i < 8; ++i) { out[i] = lo[i]; out[i + 8] = hi[i]; }
  return out;
}

// B-matrix 32x16 bf16 fed from B'[n][k] (N x K row-major, row stride ldb).
// lane<16 : col=lane,    elems 0..7 = K 0..7,   elems 8..15 = K 16..23
// lane>=16: col=lane-16, elems 0..7 = K 8..15,  elems 8..15 = K 24..31
__device__ __forceinline__ v16bf load_b_frag(const bf16_t* __restrict__ Bm, int ldb) {
  const int lane = threadIdx.x & 31;
  const int half = lane >> 4;
  const int r    = lane & 15;
  const bf16_t* base = Bm + (size_t)r * ldb + 8 * half;
  v8bf lo = *(const v8bf*)(base);
  v8bf hi = *(const v8bf*)(base + 16);
  v16bf out;
#pragma unroll
  for (int i = 0; i < 8; ++i) { out[i] = lo[i]; out[i + 8] = hi[i]; }
  return out;
}

__device__ __forceinline__ v8f wmma_bf16(v16bf a, v16bf b, v8f c) {
  return __builtin_amdgcn_wmma_f32_16x16x32_bf16(false, a, false, b,
                                                 (short)0, c, false, false);
}

// fragment bundle: 2 A-frags (32 rows) x 4 B-frags (64 cols)
struct Frag24 { v16bf a0, a1, b[4]; };

__device__ __forceinline__ void load24(Frag24& f, const bf16_t* A0,
                                       const bf16_t* A1, const bf16_t* Bp,
                                       int k, int ldab) {
  f.a0 = load_a_frag(A0 + k, ldab);
  f.a1 = load_a_frag(A1 + k, ldab);
#pragma unroll
  for (int nt = 0; nt < 4; ++nt)
    f.b[nt] = load_b_frag(Bp + (size_t)nt * 16 * ldab + k, ldab);
}

__device__ __forceinline__ void mma24(v8f acc[2][4], const Frag24& f) {
#pragma unroll
  for (int nt = 0; nt < 4; ++nt) {
    acc[0][nt] = wmma_bf16(f.a0, f.b[nt], acc[0][nt]);
    acc[1][nt] = wmma_bf16(f.a1, f.b[nt], acc[1][nt]);
  }
}

// fragment bundle: 1 A-frag (16 rows) x 4 B-frags (64 cols)
struct Frag14 { v16bf a, b[4]; };

__device__ __forceinline__ void load14(Frag14& f, const bf16_t* A,
                                       const bf16_t* Bp, int k, int ldab) {
  f.a = load_a_frag(A + k, ldab);
#pragma unroll
  for (int nt = 0; nt < 4; ++nt)
    f.b[nt] = load_b_frag(Bp + (size_t)nt * 16 * ldab + k, ldab);
}

__device__ __forceinline__ void mma14(v8f acc[4], const Frag14& f) {
#pragma unroll
  for (int nt = 0; nt < 4; ++nt) acc[nt] = wmma_bf16(f.a, f.b[nt], acc[nt]);
}

// ---- elementwise f32 -> bf16 ----------------------------------------------
__global__ void f32_to_bf16_kernel(const float* __restrict__ src,
                                   bf16_t* __restrict__ dst, int n) {
  int i = blockIdx.x * blockDim.x + threadIdx.x;
  if (i < n) dst[i] = (bf16_t)src[i];
}

// ---- QKV projection: C[8192,3072] = xb[8192,1024] @ wqkv[3072,1024]^T ------
// epilogue: cols [0,1024)   -> q  = elu(c)+1, layout [B,T,H,QK]
//           cols [1024,2048)-> kT = elu(c)+1, layout [B,H,QK,T] (time-major)
//           cols [2048,3072)-> vT = c,        layout [B,H,VD,T]
__global__ __launch_bounds__(128) void qkv_gemm_kernel(
    const bf16_t* __restrict__ xb, const bf16_t* __restrict__ wb,
    bf16_t* __restrict__ qb, bf16_t* __restrict__ kT, bf16_t* __restrict__ vT) {
  const int K    = D_;
  const int wave = threadIdx.x >> 5;
  const int row0 = blockIdx.x * 128 + wave * 32;   // two 16-row frags
  const int col0 = blockIdx.y * 64;

  const bf16_t* A0 = xb + (size_t)row0 * K;
  const bf16_t* A1 = A0 + (size_t)16 * K;
  const bf16_t* Bp = wb + (size_t)col0 * K;

  v8f acc[2][4] = {};

  // fully unrolled: 32 independent k-steps, scheduler pipelines the loads
#pragma unroll
  for (int ks = 0; ks < D_ / 32; ++ks) {
    Frag24 f;
    load24(f, A0, A1, Bp, ks * 32, K);
    mma24(acc, f);
  }

  const int lane = threadIdx.x & 31;
  const int half = lane >> 4;
  const int cl   = lane & 15;
#pragma unroll
  for (int mt = 0; mt < 2; ++mt) {
#pragma unroll
    for (int nt = 0; nt < 4; ++nt) {
#pragma unroll
      for (int i = 0; i < 8; ++i) {
        const int m = row0 + mt * 16 + i + 8 * half;   // global row = b*T + t
        const int n = col0 + nt * 16 + cl;             // global col
        float v = acc[mt][nt][i];
        if (n < 2048) v = (v > 0.f) ? (v + 1.f) : __expf(v);  // elu(x)+1
        const bf16_t bv = (bf16_t)v;
        if (n < 1024) {
          qb[(size_t)m * (H_ * QK_) + n] = bv;
        } else {
          const int local = n & 1023;
          const int h = local >> 6;
          const int d = local & 63;
          const int b = m >> 12;                       // T_ = 4096
          const int t = m & (T_ - 1);
          const size_t idx = (((size_t)b * H_ + h) * 64 + d) * (size_t)T_ + t;
          if (n < 2048) kT[idx] = bv; else vT[idx] = bv;
        }
      }
    }
  }
}

// ---- kv partials: kv_part[bh,ks][d][m] = sum_{t in slice ks} kT[d,t]*vT[m,t]
// K (time) split 8 ways for parallelism; deterministic f32 partial stores.
__global__ __launch_bounds__(128) void kv_gemm_kernel(
    const bf16_t* __restrict__ kT, const bf16_t* __restrict__ vT,
    float* __restrict__ kv_part) {
  const int bh   = blockIdx.x >> 3;                // b*H + h
  const int ksl  = blockIdx.x & 7;                 // k-slice
  const int kbeg = ksl * (T_ / 8);
  const int wave = threadIdx.x >> 5;
  const int row0 = wave * 16;                      // d band
  const bf16_t* A  = kT + (size_t)bh * 64 * T_ + (size_t)row0 * T_ + kbeg;
  const bf16_t* Bm = vT + (size_t)bh * 64 * T_ + kbeg;
  float* C = kv_part + (size_t)blockIdx.x * 64 * 64;

  v8f acc[4] = {};
  // fully unrolled: 16 independent k-steps over the 512-wide slice
#pragma unroll
  for (int ks = 0; ks < (T_ / 8) / 32; ++ks) {
    Frag14 f;
    load14(f, A, Bm, ks * 32, T_);
    mma14(acc, f);
  }

  const int lane = threadIdx.x & 31;
  const int half = lane >> 4;
  const int cl   = lane & 15;
#pragma unroll
  for (int nt = 0; nt < 4; ++nt) {
#pragma unroll
    for (int i = 0; i < 8; ++i) {
      const int d = row0 + i + 8 * half;
      const int m = nt * 16 + cl;
      C[(size_t)d * 64 + m] = acc[nt][i];
    }
  }
}

// ---- reduce 8 kv partials -> bf16 kvT[b,h][m][d] (transposed operand) ------
__global__ __launch_bounds__(256) void kv_reduce_kernel(
    const float* __restrict__ kv_part, bf16_t* __restrict__ kvT) {
  const int bh = blockIdx.x;
  for (int idx = threadIdx.x; idx < 4096; idx += 256) {
    const int d = idx >> 6;
    const int m = idx & 63;
    float s = 0.f;
#pragma unroll
    for (int p = 0; p < 8; ++p)
      s += kv_part[(((size_t)bh * 8 + p) * 64 + d) * 64 + m];
    kvT[(size_t)bh * 80 * 64 + (size_t)m * 64 + d] = (bf16_t)s;
  }
}

// ---- ksum[b,h,d] = sum_t kT[b,h,d,t] -> row 64 of kvT; rows 65..79 = 0 -----
__global__ __launch_bounds__(64) void ksum_kernel(
    const bf16_t* __restrict__ kT, bf16_t* __restrict__ kvT) {
  const int bh = blockIdx.x;
  const int d  = threadIdx.x;                      // 0..63
  const bf16_t* row = kT + ((size_t)bh * 64 + d) * T_;
  float s = 0.f;
  for (int t = 0; t < T_; t += 8) {
    v8bf v = *(const v8bf*)(row + t);
#pragma unroll
    for (int j = 0; j < 8; ++j) s += (float)v[j];
  }
  bf16_t* base = kvT + (size_t)bh * 80 * 64;
  base[64 * 64 + d] = (bf16_t)s;
#pragma unroll
  for (int rr = 65; rr < 80; ++rr) base[(size_t)rr * 64 + d] = (bf16_t)0.f;
}

// ---- o[b,t,h,:] = (q . kv) * 1/(q . ksum) ; N=80 (5th tile col0 = q.ksum) --
__global__ __launch_bounds__(128) void o_gemm_kernel(
    const bf16_t* __restrict__ qb, const bf16_t* __restrict__ kvT,
    bf16_t* __restrict__ ob) {
  const int blk  = blockIdx.x;
  const int tblk = blk & 31;                       // T/128 = 32 tiles
  const int bh   = blk >> 5;
  const int b    = bh >> 4;
  const int h    = bh & 15;
  const int wave = threadIdx.x >> 5;
  const int t0   = tblk * 128 + wave * 32;

  const bf16_t* A0 = qb + (((size_t)b * T_ + t0) * H_ + h) * QK_;  // row stride H*QK
  const bf16_t* A1 = A0 + (size_t)16 * (H_ * QK_);
  const bf16_t* Bm = kvT + (size_t)bh * 80 * 64;

  v8f acc[2][5] = {};
  // K = 64: two k-steps, fully unrolled
#pragma unroll
  for (int ks = 0; ks < 2; ++ks) {
    v16bf a0 = load_a_frag(A0 + ks * 32, H_ * QK_);
    v16bf a1 = load_a_frag(A1 + ks * 32, H_ * QK_);
    v16bf bfr[5];
#pragma unroll
    for (int nt = 0; nt < 5; ++nt)
      bfr[nt] = load_b_frag(Bm + (size_t)nt * 16 * 64 + ks * 32, 64);
#pragma unroll
    for (int nt = 0; nt < 5; ++nt) {
      acc[0][nt] = wmma_bf16(a0, bfr[nt], acc[0][nt]);
      acc[1][nt] = wmma_bf16(a1, bfr[nt], acc[1][nt]);
    }
  }

  const int lane = threadIdx.x & 31;
  const int half = lane >> 4;
  const int cl   = lane & 15;
  // broadcast q.ksum (tile 4, col 0 -> lane half*16) to the whole half-wave
  float inv[2][8];
#pragma unroll
  for (int mt = 0; mt < 2; ++mt) {
#pragma unroll
    for (int i = 0; i < 8; ++i) {
      int bits = __builtin_amdgcn_ds_bpermute((half * 16) * 4,
                                              __float_as_int(acc[mt][4][i]));
      inv[mt][i] = 1.0f / __int_as_float(bits);
    }
  }
#pragma unroll
  for (int mt = 0; mt < 2; ++mt) {
#pragma unroll
    for (int nt = 0; nt < 4; ++nt) {
#pragma unroll
      for (int i = 0; i < 8; ++i) {
        const int t = t0 + mt * 16 + i + 8 * half;
        const int m = nt * 16 + cl;
        ob[(((size_t)b * T_ + t) * H_ + h) * VD_ + m] =
            (bf16_t)(acc[mt][nt][i] * inv[mt][i]);
      }
    }
  }
}

// ---- out[b,t,h,c] = sum_m o[b,t,h,m] * Wproj[c,m]  (M=131072,N=1024,K=64) --
__global__ __launch_bounds__(128) void proj_gemm_kernel(
    const bf16_t* __restrict__ ob, const bf16_t* __restrict__ wpb,
    float* __restrict__ out) {
  const int wave = threadIdx.x >> 5;
  const int row0 = blockIdx.x * 128 + wave * 32;   // row = (b,t,h) flat
  const int col0 = blockIdx.y * 64;
  const bf16_t* A0 = ob + (size_t)row0 * VD_;
  const bf16_t* A1 = A0 + (size_t)16 * VD_;
  const bf16_t* Bp = wpb + (size_t)col0 * VD_;

  v8f acc[2][4] = {};
#pragma unroll
  for (int ks = 0; ks < 2; ++ks) {
    Frag24 f;
    load24(f, A0, A1, Bp, ks * 32, VD_);
    mma24(acc, f);
  }

  const int lane = threadIdx.x & 31;
  const int half = lane >> 4;
  const int cl   = lane & 15;
#pragma unroll
  for (int mt = 0; mt < 2; ++mt) {
#pragma unroll
    for (int nt = 0; nt < 4; ++nt) {
#pragma unroll
      for (int i = 0; i < 8; ++i) {
        const size_t m = (size_t)row0 + mt * 16 + i + 8 * half;
        out[m * D_ + col0 + nt * 16 + cl] = acc[mt][nt][i];
      }
    }
  }
}

// ---------------------------------------------------------------------------
extern "C" void kernel_launch(void* const* d_in, const int* in_sizes, int n_in,
                              void* d_out, int out_size, void* d_ws, size_t ws_size,
                              hipStream_t stream) {
  const float* x     = (const float*)d_in[0];   // [2,4096,1024]
  const float* WQ    = (const float*)d_in[1];   // [1024,1024]
  const float* WK    = (const float*)d_in[2];
  const float* WV    = (const float*)d_in[3];
  const float* Wproj = (const float*)d_in[4];   // [1024,64]
  float* out = (float*)d_out;                   // [2,4096,16,1024]

  bf16_t* ws = (bf16_t*)d_ws;
  // workspace layout (bf16 elements; f32 partials appended at the end)
  const size_t N_X    = (size_t)B_ * T_ * D_;          // 8,388,608
  const size_t N_WQKV = (size_t)3 * D_ * D_;           // 3,145,728
  const size_t N_WP   = (size_t)D_ * VD_;              // 65,536
  const size_t N_QKV  = (size_t)B_ * T_ * H_ * 64;     // 8,388,608 each
  const size_t N_KVT  = (size_t)B_ * H_ * 80 * 64;     // 163,840

  bf16_t* xb    = ws;
  bf16_t* wqkvb = xb + N_X;
  bf16_t* wpb   = wqkvb + N_WQKV;
  bf16_t* qb    = wpb + N_WP;
  bf16_t* kT    = qb + N_QKV;
  bf16_t* vT    = kT + N_QKV;
  bf16_t* kvT   = vT + N_QKV;
  float*  kv_part = (float*)(kvT + N_KVT);      // 256*64*64 f32 = 4 MB
  bf16_t* ob    = xb;   // xb is dead after qkv_gemm; reuse (rewritten every call)
  (void)in_sizes; (void)n_in; (void)out_size; (void)ws_size;

  // 1) convert everything to bf16
  f32_to_bf16_kernel<<<(int)((N_X + 255) / 256), 256, 0, stream>>>(x, xb, (int)N_X);
  f32_to_bf16_kernel<<<(D_ * D_ + 255) / 256, 256, 0, stream>>>(WQ, wqkvb, D_ * D_);
  f32_to_bf16_kernel<<<(D_ * D_ + 255) / 256, 256, 0, stream>>>(WK, wqkvb + (size_t)D_ * D_, D_ * D_);
  f32_to_bf16_kernel<<<(D_ * D_ + 255) / 256, 256, 0, stream>>>(WV, wqkvb + (size_t)2 * D_ * D_, D_ * D_);
  f32_to_bf16_kernel<<<(D_ * VD_ + 255) / 256, 256, 0, stream>>>(Wproj, wpb, D_ * VD_);

  // 2) fused QKV projection + feature map (+ time-major k/v)
  qkv_gemm_kernel<<<dim3((B_ * T_) / 128, 3072 / 64), 128, 0, stream>>>(xb, wqkvb, qb, kT, vT);

  // 3) per-head kv = K^T V (8-way K-split partials), reduce, ksum row
  kv_gemm_kernel<<<B_ * H_ * 8, 128, 0, stream>>>(kT, vT, kv_part);
  kv_reduce_kernel<<<B_ * H_, 256, 0, stream>>>(kv_part, kvT);
  ksum_kernel<<<B_ * H_, 64, 0, stream>>>(kT, kvT);

  // 4) o = (q . kv) normalized
  o_gemm_kernel<<<B_ * H_ * (T_ / 128), 128, 0, stream>>>(qb, kvT, ob);

  // 5) per-head output projection -> f32 out
  proj_gemm_kernel<<<dim3((B_ * T_ * H_) / 128, D_ / 64), 128, 0, stream>>>(ob, wpb, out);
}
